// MultiHeadLSRAttentionTriton_67499706024042
// MI455X (gfx1250) — matrix operand order repaired and checked by reading, not compiled, not run on hardware
//
#include <hip/hip_runtime.h>

// ---------------------------------------------------------------------------
// MI455X (gfx1250) bf16-WMMA implementation of multi-head LSR attention.
// Wave32; all matrix math via v_wmma_f32_16x16x32_bf16 (f32 accumulation).
// GEMMs: 32x64 per wave, LDS double-buffered B tiles via async-to-LDS copies.
// ---------------------------------------------------------------------------

typedef __bf16 bf16_t;
typedef __attribute__((ext_vector_type(16))) __bf16 v16bf;
typedef __attribute__((ext_vector_type(8)))  float  v8f;

#define D_MODEL 1024
#define N_HEADS 16
#define D_HEAD  64
#define LSR_R   32
#define BATCH   2
#define SEQ     2048
#define BT      (BATCH * SEQ)   // 4096

#if defined(__AMDGCN__) && __has_builtin(__builtin_amdgcn_global_load_async_to_lds_b128)
#define HAVE_ASYNC_LDS 1
#else
#define HAVE_ASYNC_LDS 0
#endif

#if HAVE_ASYNC_LDS
// Builtin parameter types (from hipcc diagnostic): non-const pointers to
// int __attribute__((vector_size(16))) in global (AS1) / LDS (AS3).
typedef int v4i_vec __attribute__((vector_size(16)));
typedef __attribute__((address_space(1))) v4i_vec* as1_v4i_p;
typedef __attribute__((address_space(3))) v4i_vec* as3_v4i_p;
#endif

// ---- WMMA wrapper ---------------------------------------------------------
__device__ __forceinline__ v8f wmma_bf16(v16bf a, v16bf b, v8f c) {
  // (neg_a, A, neg_b, B, c_mod, C, reuse_a, reuse_b)
  return __builtin_amdgcn_wmma_f32_16x16x32_bf16(false, a, false, b,
                                                 (short)0, c, false, false);
}

// ---- Fragment loaders (layouts per CDNA5 ISA 7.12.2, wave32) --------------
// A-matrix 16x32 bf16, row-major source (K contiguous).
// lanes 0-15: row=lane,    K = {0..7, 16..23}
// lanes16-31: row=lane-16, K = {8..15, 24..31}
__device__ __forceinline__ v16bf load_fragA_rowK(const bf16_t* __restrict__ base,
                                                 int ld) {
  const int lane = threadIdx.x & 31;
  const bf16_t* p = base + (size_t)(lane & 15) * ld + ((lane < 16) ? 0 : 8);
  union { uint4 u[2]; v16bf v; } u;
  u.u[0] = *(const uint4*)p;         // K offsets +0..+7
  u.u[1] = *(const uint4*)(p + 16);  // K offsets +16..+23
  return u.v;
}

// B-matrix 32x16 bf16 with K-contiguous storage laid out as [n][k], stride ld.
// lane n=lane&15; lanes 0-15 hold K={0..7,16..23}, lanes 16-31 hold K={8..15,24..31}.
__device__ __forceinline__ v16bf load_fragB_colK(const bf16_t* __restrict__ base,
                                                 int ld) {
  const int lane = threadIdx.x & 31;
  const bf16_t* p = base + (size_t)(lane & 15) * ld + ((lane < 16) ? 0 : 8);
  union { uint4 u[2]; v16bf v; } u;
  u.u[0] = *(const uint4*)p;
  u.u[1] = *(const uint4*)(p + 16);
  return u.v;
}

// ---- Async LDS staging helpers --------------------------------------------
__device__ __forceinline__ void gemm_wait_copy() {
#if HAVE_ASYNC_LDS
#if __has_builtin(__builtin_amdgcn_s_wait_asynccnt)
  __builtin_amdgcn_s_wait_asynccnt(0);
#else
  asm volatile("s_wait_asynccnt 0x0" ::: "memory");
#endif
#endif
}

// Copy one 64x32 bf16 B tile (4KB) from global (row stride D_MODEL) into LDS
// (row stride 32). 128 threads, 32B per thread.
__device__ __forceinline__ void copy_Btile(const bf16_t* __restrict__ src,
                                           bf16_t* __restrict__ dstLds) {
  const int tid = threadIdx.x;
  const int n    = tid >> 1;
  const int koff = (tid & 1) * 16;
  const bf16_t* g = src + (size_t)n * D_MODEL + koff;
  bf16_t* l = dstLds + n * 32 + koff;
#if HAVE_ASYNC_LDS
  __builtin_amdgcn_global_load_async_to_lds_b128((as1_v4i_p)g, (as3_v4i_p)l, 0, 0);
  __builtin_amdgcn_global_load_async_to_lds_b128((as1_v4i_p)(g + 8),
                                                 (as3_v4i_p)(l + 8), 0, 0);
#else
  uint4 t0 = *(const uint4*)g;
  uint4 t1 = *(const uint4*)(g + 8);
  *(uint4*)l = t0;
  *(uint4*)(l + 8) = t1;
#endif
}

// ---- Conversion / transpose kernels ---------------------------------------
__global__ void k_cvt_bf16(const float* __restrict__ X, bf16_t* __restrict__ Y,
                           int total) {
  int idx = blockIdx.x * 256 + threadIdx.x;
  if (idx < total) Y[idx] = (bf16_t)X[idx];
}

// W: 1024x1024 row-major (K x N)  ->  WT: N x K row-major (bf16)
__global__ void k_cvt_transpose_w(const float* __restrict__ W,
                                  bf16_t* __restrict__ WT) {
  int idx = blockIdx.x * 256 + threadIdx.x;   // 1<<20 total
  int k = idx >> 10, n = idx & 1023;
  WT[((size_t)n << 10) + k] = (bf16_t)W[idx];
}

// Wlsr: (H, Dh, R) -> (H, R, Dh) bf16
__global__ void k_cvt_transpose_lsr(const float* __restrict__ W,
                                    bf16_t* __restrict__ WT) {
  int idx = blockIdx.x * 256 + threadIdx.x;   // 16*64*32 = 32768 total
  int h = idx >> 11;
  int rem = idx & 2047;
  int d = rem >> 5, r = rem & 31;
  WT[((size_t)h * LSR_R + r) * D_HEAD + d] = (bf16_t)W[idx];
}

// ---- Fused QKV projection GEMM --------------------------------------------
// C(4096x1024) = Xb @ W + bias ; z selects {Q, K, V}.
// Each wave: 32 rows x 64 cols. B tile staged in LDS, double buffered.
// Q,K stored (B,H,T,Dh); V stored transposed (B,H,Dh,T).
__global__ __launch_bounds__(128)
void k_gemm_qkv(const bf16_t* __restrict__ Xb,
                const bf16_t* __restrict__ WqT, const bf16_t* __restrict__ WkT,
                const bf16_t* __restrict__ WvT,
                const float* __restrict__ bq, const float* __restrict__ bk,
                const float* __restrict__ bv,
                bf16_t* __restrict__ Q, bf16_t* __restrict__ Kh,
                bf16_t* __restrict__ Vt) {
  __shared__ __align__(16) bf16_t Bs[2][64 * 32];
  const int wave = threadIdx.x >> 5;
  const int lane = threadIdx.x & 31;
  const int m0 = blockIdx.y * 128 + wave * 32;
  const int n0 = blockIdx.x * 64;
  const int which = blockIdx.z;
  const bf16_t* WT  = (which == 0) ? WqT : (which == 1) ? WkT : WvT;
  const float* bias = (which == 0) ? bq  : (which == 1) ? bk  : bv;
  const bf16_t* Bt    = WT + (size_t)n0 * D_MODEL;
  const bf16_t* Abase = Xb + (size_t)m0 * D_MODEL;

  v8f acc[2][4] = {};
  copy_Btile(Bt, Bs[0]);
  for (int ks = 0; ks < D_MODEL / 32; ++ks) {
    const int buf = ks & 1;
    gemm_wait_copy();
    __syncthreads();
    if (ks + 1 < D_MODEL / 32) copy_Btile(Bt + (ks + 1) * 32, Bs[buf ^ 1]);
    v16bf a0 = load_fragA_rowK(Abase + ks * 32, D_MODEL);
    v16bf a1 = load_fragA_rowK(Abase + 16 * D_MODEL + ks * 32, D_MODEL);
#pragma unroll
    for (int j = 0; j < 4; ++j) {
      v16bf b = load_fragB_colK(&Bs[buf][16 * j * 32], 32);
      acc[0][j] = wmma_bf16(a0, b, acc[0][j]);
      acc[1][j] = wmma_bf16(a1, b, acc[1][j]);
    }
  }

  const int col = lane & 15;
  const int rb  = (lane < 16) ? 0 : 8;
#pragma unroll
  for (int r = 0; r < 2; ++r) {
#pragma unroll
    for (int j = 0; j < 4; ++j) {
      int n = n0 + 16 * j + col;
      int h = n >> 6, d = n & 63;
      float bia = bias[n];
#pragma unroll
      for (int v = 0; v < 8; ++v) {
        int m = m0 + r * 16 + rb + v;
        int bb = m >> 11, t = m & (SEQ - 1);
        bf16_t val = (bf16_t)(acc[r][j][v] + bia);
        if (which == 0)
          Q [(((size_t)bb * N_HEADS + h) * SEQ + t) * D_HEAD + d] = val;
        else if (which == 1)
          Kh[(((size_t)bb * N_HEADS + h) * SEQ + t) * D_HEAD + d] = val;
        else
          Vt[(((size_t)bb * N_HEADS + h) * D_HEAD + d) * SEQ + t] = val;
      }
    }
  }
}

// ---- Low-rank projections: q_lr = Q_head @ Wq_lsr_head --------------------
// (T x 64) @ (64 x 32) per (b,h). z selects Q-path vs K-path.
__global__ __launch_bounds__(128)
void k_lr_proj(const bf16_t* __restrict__ Q, const bf16_t* __restrict__ Kh,
               const bf16_t* __restrict__ WlqT, const bf16_t* __restrict__ WlkT,
               bf16_t* __restrict__ qlr, bf16_t* __restrict__ klr) {
  const int wave = threadIdx.x >> 5;
  const int lane = threadIdx.x & 31;
  const int t0 = blockIdx.x * 64 + wave * 16;
  const int bh = blockIdx.y;           // 0..31
  const int h  = bh & (N_HEADS - 1);
  const bf16_t* src = (blockIdx.z == 0) ? Q    : Kh;
  const bf16_t* Wl  = (blockIdx.z == 0) ? WlqT : WlkT;
  bf16_t*       dst = (blockIdx.z == 0) ? qlr  : klr;

  const bf16_t* Abase = src + ((size_t)bh * SEQ + t0) * D_HEAD;
  const bf16_t* Bbase = Wl + (size_t)h * LSR_R * D_HEAD;

  v8f acc0 = {}, acc1 = {};
#pragma unroll
  for (int kk = 0; kk < D_HEAD; kk += 32) {
    v16bf a  = load_fragA_rowK(Abase + kk, D_HEAD);
    v16bf b0 = load_fragB_colK(Bbase + 0 * D_HEAD + kk, D_HEAD);
    v16bf b1 = load_fragB_colK(Bbase + 16 * D_HEAD + kk, D_HEAD);
    acc0 = wmma_bf16(a, b0, acc0);
    acc1 = wmma_bf16(a, b1, acc1);
  }
  const int col = lane & 15;
  const int rb  = (lane < 16) ? 0 : 8;
#pragma unroll
  for (int v = 0; v < 8; ++v) {
    bf16_t* o = dst + ((size_t)bh * SEQ + (t0 + rb + v)) * LSR_R;
    o[col]      = (bf16_t)acc0[v];
    o[col + 16] = (bf16_t)acc1[v];
  }
}

// ---- Flash-style causal attention on low-rank scores ----------------------
// One wave owns 16 query rows; kv processed in tiles of 32.
// scores tile = two WMMAs (K dim == R == 32 exactly); P@V = 4 WMMAs per tile.
// Next tile's K fragments and current V fragments are prefetched before the
// softmax VALU block so global loads overlap exp/shuffle work.
__global__ __launch_bounds__(128)
void k_attn(const bf16_t* __restrict__ qlr, const bf16_t* __restrict__ klr,
            const bf16_t* __restrict__ Vt, bf16_t* __restrict__ attn_out) {
  __shared__ __align__(16) bf16_t Pst[4][16 * 32];  // per-wave P staging
  const int wave = threadIdx.x >> 5;
  const int lane = threadIdx.x & 31;
  const int bh = blockIdx.y;
  const int q0 = (blockIdx.x * 4 + wave) * 16;
  const int col = lane & 15;
  const int rb  = (lane < 16) ? 0 : 8;
  const float scale = 0.17677669529663687f;   // 1/sqrt(R)

  v8f acc[4] = {};
  float mi[8], li[8];
#pragma unroll
  for (int v = 0; v < 8; ++v) { mi[v] = -__builtin_inff(); li[v] = 0.f; }

  const bf16_t* Kbase = klr + (size_t)bh * SEQ * LSR_R;
  const bf16_t* Vbase = Vt + (size_t)bh * D_HEAD * SEQ;
  const v16bf aq = load_fragA_rowK(qlr + ((size_t)bh * SEQ + q0) * LSR_R, LSR_R);
  bf16_t* P = &Pst[wave][0];
  const v8f zero = {};

  v16bf bk0 = load_fragB_colK(Kbase, LSR_R);
  v16bf bk1 = load_fragB_colK(Kbase + 16 * LSR_R, LSR_R);

  for (int j = 0; j < q0 + 16; j += 32) {
    v8f s0 = wmma_bf16(aq, bk0, zero);
    v8f s1 = wmma_bf16(aq, bk1, zero);

    // prefetch next kv tile's K fragments (overlaps softmax below)
    if (j + 32 < q0 + 16) {
      bk0 = load_fragB_colK(Kbase + (size_t)(j + 32) * LSR_R, LSR_R);
      bk1 = load_fragB_colK(Kbase + (size_t)(j + 48) * LSR_R, LSR_R);
    }
    // preload this tile's V fragments (independent of P)
    v16bf bv0 = load_fragB_colK(Vbase + (size_t)(0 * 16) * SEQ + j, SEQ);
    v16bf bv1 = load_fragB_colK(Vbase + (size_t)(1 * 16) * SEQ + j, SEQ);
    v16bf bv2 = load_fragB_colK(Vbase + (size_t)(2 * 16) * SEQ + j, SEQ);
    v16bf bv3 = load_fragB_colK(Vbase + (size_t)(3 * 16) * SEQ + j, SEQ);

    const bool diag = (j + 32 > q0);
    float p0[8], p1[8];
#pragma unroll
    for (int v = 0; v < 8; ++v) {
      float x0 = s0[v] * scale;
      float x1 = s1[v] * scale;
      const int row = q0 + rb + v;
      if (diag) {
        if (j + col > row)      x0 = -__builtin_inff();
        if (j + 16 + col > row) x1 = -__builtin_inff();
      }
      // row max across the 16-lane half (rows live in lanes sharing bit4)
      float rm = fmaxf(x0, x1);
      rm = fmaxf(rm, __shfl_xor(rm, 1, 32));
      rm = fmaxf(rm, __shfl_xor(rm, 2, 32));
      rm = fmaxf(rm, __shfl_xor(rm, 4, 32));
      rm = fmaxf(rm, __shfl_xor(rm, 8, 32));
      float mnew  = fmaxf(mi[v], rm);
      float alpha = __expf(mi[v] - mnew);
      float e0 = __expf(x0 - mnew);
      float e1 = __expf(x1 - mnew);
      float rs = e0 + e1;
      rs += __shfl_xor(rs, 1, 32);
      rs += __shfl_xor(rs, 2, 32);
      rs += __shfl_xor(rs, 4, 32);
      rs += __shfl_xor(rs, 8, 32);
      li[v] = li[v] * alpha + rs;
      mi[v] = mnew;
      p0[v] = e0; p1[v] = e1;
#pragma unroll
      for (int d = 0; d < 4; ++d) acc[d][v] *= alpha;
    }

    // Transpose P (C-layout -> A-layout) through LDS, then 4 P@V WMMAs.
#pragma unroll
    for (int v = 0; v < 8; ++v) {
      P[(rb + v) * 32 + col]      = (bf16_t)p0[v];
      P[(rb + v) * 32 + col + 16] = (bf16_t)p1[v];
    }
    v16bf ap = load_fragA_rowK(P, 32);   // in-wave LDS: DScnt keeps order
    acc[0] = wmma_bf16(ap, bv0, acc[0]);
    acc[1] = wmma_bf16(ap, bv1, acc[1]);
    acc[2] = wmma_bf16(ap, bv2, acc[2]);
    acc[3] = wmma_bf16(ap, bv3, acc[3]);
  }

  const int bb = bh >> 4, h = bh & (N_HEADS - 1);
#pragma unroll
  for (int v = 0; v < 8; ++v) {
    float inv = 1.0f / li[v];
    int t = q0 + rb + v;
    bf16_t* o = attn_out + ((size_t)bb * SEQ + t) * D_MODEL + h * D_HEAD;
#pragma unroll
    for (int d = 0; d < 4; ++d)
      o[d * 16 + col] = (bf16_t)(acc[d][v] * inv);
  }
}

// ---- Output projection: out = attn @ Wo + bo (f32 result) -----------------
__global__ __launch_bounds__(128)
void k_gemm_out(const bf16_t* __restrict__ Ab, const bf16_t* __restrict__ WoT,
                const float* __restrict__ bo, float* __restrict__ out) {
  __shared__ __align__(16) bf16_t Bs[2][64 * 32];
  const int wave = threadIdx.x >> 5;
  const int lane = threadIdx.x & 31;
  const int m0 = blockIdx.y * 128 + wave * 32;
  const int n0 = blockIdx.x * 64;
  const bf16_t* Bt    = WoT + (size_t)n0 * D_MODEL;
  const bf16_t* Abase = Ab + (size_t)m0 * D_MODEL;

  v8f acc[2][4] = {};
  copy_Btile(Bt, Bs[0]);
  for (int ks = 0; ks < D_MODEL / 32; ++ks) {
    const int buf = ks & 1;
    gemm_wait_copy();
    __syncthreads();
    if (ks + 1 < D_MODEL / 32) copy_Btile(Bt + (ks + 1) * 32, Bs[buf ^ 1]);
    v16bf a0 = load_fragA_rowK(Abase + ks * 32, D_MODEL);
    v16bf a1 = load_fragA_rowK(Abase + 16 * D_MODEL + ks * 32, D_MODEL);
#pragma unroll
    for (int j = 0; j < 4; ++j) {
      v16bf b = load_fragB_colK(&Bs[buf][16 * j * 32], 32);
      acc[0][j] = wmma_bf16(a0, b, acc[0][j]);
      acc[1][j] = wmma_bf16(a1, b, acc[1][j]);
    }
  }

  const int col = lane & 15;
  const int rb  = (lane < 16) ? 0 : 8;
#pragma unroll
  for (int r = 0; r < 2; ++r) {
#pragma unroll
    for (int j = 0; j < 4; ++j) {
      int n = n0 + 16 * j + col;
      float bia = bo[n];
#pragma unroll
      for (int v = 0; v < 8; ++v)
        out[(size_t)(m0 + r * 16 + rb + v) * D_MODEL + n] = acc[r][j][v] + bia;
    }
  }
}

// ---------------------------------------------------------------------------
extern "C" void kernel_launch(void* const* d_in, const int* in_sizes, int n_in,
                              void* d_out, int out_size, void* d_ws, size_t ws_size,
                              hipStream_t stream) {
  const float* x      = (const float*)d_in[0];
  const float* Wq     = (const float*)d_in[1];
  const float* bq     = (const float*)d_in[2];
  const float* Wk     = (const float*)d_in[3];
  const float* bk     = (const float*)d_in[4];
  const float* Wv     = (const float*)d_in[5];
  const float* bv     = (const float*)d_in[6];
  const float* Wo     = (const float*)d_in[7];
  const float* bo     = (const float*)d_in[8];
  const float* Wqlsr  = (const float*)d_in[9];
  const float* Wklsr  = (const float*)d_in[10];
  float* out = (float*)d_out;

  // ---- workspace carve-up (256B aligned) ----
  char* ws = (char*)d_ws;
  size_t off = 0;
  auto take = [&](size_t elems) {
    char* p = ws + off;
    off += (elems * sizeof(bf16_t) + 255) & ~(size_t)255;
    return (bf16_t*)p;
  };
  bf16_t* xbf  = take((size_t)BT * D_MODEL);
  bf16_t* WqT  = take((size_t)D_MODEL * D_MODEL);
  bf16_t* WkT  = take((size_t)D_MODEL * D_MODEL);
  bf16_t* WvT  = take((size_t)D_MODEL * D_MODEL);
  bf16_t* WoT  = take((size_t)D_MODEL * D_MODEL);
  bf16_t* WlqT = take((size_t)N_HEADS * LSR_R * D_HEAD);
  bf16_t* WlkT = take((size_t)N_HEADS * LSR_R * D_HEAD);
  bf16_t* Q    = take((size_t)BT * D_MODEL);
  bf16_t* Kh   = take((size_t)BT * D_MODEL);
  bf16_t* Vt   = take((size_t)BT * D_MODEL);
  bf16_t* qlr  = take((size_t)BATCH * N_HEADS * SEQ * LSR_R);
  bf16_t* klr  = take((size_t)BATCH * N_HEADS * SEQ * LSR_R);
  bf16_t* attn = take((size_t)BT * D_MODEL);
  (void)ws_size; (void)in_sizes; (void)n_in; (void)out_size;

  // 1) conversions / transposes
  k_cvt_bf16<<<(BT * D_MODEL + 255) / 256, 256, 0, stream>>>(x, xbf, BT * D_MODEL);
  k_cvt_transpose_w<<<4096, 256, 0, stream>>>(Wq, WqT);
  k_cvt_transpose_w<<<4096, 256, 0, stream>>>(Wk, WkT);
  k_cvt_transpose_w<<<4096, 256, 0, stream>>>(Wv, WvT);
  k_cvt_transpose_w<<<4096, 256, 0, stream>>>(Wo, WoT);
  k_cvt_transpose_lsr<<<128, 256, 0, stream>>>(Wqlsr, WlqT);
  k_cvt_transpose_lsr<<<128, 256, 0, stream>>>(Wklsr, WlkT);

  // 2) fused QKV projection (z = q/k/v)
  k_gemm_qkv<<<dim3(D_MODEL / 64, BT / 128, 3), 128, 0, stream>>>(
      xbf, WqT, WkT, WvT, bq, bk, bv, Q, Kh, Vt);

  // 3) low-rank projections (z = q/k)
  k_lr_proj<<<dim3(SEQ / 64, BATCH * N_HEADS, 2), 128, 0, stream>>>(
      Q, Kh, WlqT, WlkT, qlr, klr);

  // 4) causal flash attention over low-rank scores
  k_attn<<<dim3(SEQ / 64, BATCH * N_HEADS), 128, 0, stream>>>(qlr, klr, Vt, attn);

  // 5) output projection -> f32
  k_gemm_out<<<dim3(D_MODEL / 64, BT / 128), 128, 0, stream>>>(attn, WoT, bo, out);
}
